// GROUP_MIL_40827959115966
// MI455X (gfx1250) — matrix operand order repaired and checked by reading, not compiled, not run on hardware
//
#include <hip/hip_runtime.h>
#include <math.h>

#define N_INST 100000
#define D_IN   1024
#define DD     512
#define NG     8
#define GS     12500
#define NC     2

typedef __attribute__((ext_vector_type(16))) __bf16 v16bf;
typedef __attribute__((ext_vector_type(2)))  __bf16 v2bf;
typedef __attribute__((ext_vector_type(8)))  float  v8f;

// scheduling-group pins (no-op if builtin unavailable)
#if __has_builtin(__builtin_amdgcn_sched_group_barrier)
#define SGB(mask, size, id) __builtin_amdgcn_sched_group_barrier(mask, size, id)
#else
#define SGB(mask, size, id)
#endif
#define SGB_WMMA   0x008
#define SGB_VMREAD 0x020
#define SGB_DSREAD 0x100

struct U32x8 { uint4 lo; uint4 hi; };

static __device__ __forceinline__ unsigned short f2bf(float f) {
    unsigned u = __float_as_uint(f);
    unsigned r = u + 0x7FFFu + ((u >> 16) & 1u);   // round-to-nearest-even
    return (unsigned short)(r >> 16);
}

// pack two f32 -> two bf16 in one dword (HW packed convert if available)
static __device__ __forceinline__ unsigned pk2bf(float a, float b) {
#if __has_builtin(__builtin_amdgcn_cvt_pk_bf16_f32)
    v2bf p = __builtin_amdgcn_cvt_pk_bf16_f32(a, b);
    return __builtin_bit_cast(unsigned, p);
#else
    return (unsigned)f2bf(a) | ((unsigned)f2bf(b) << 16);
#endif
}

static __device__ __forceinline__ v16bf frag_lds(const unsigned short* lo,
                                                 const unsigned short* hi) {
    U32x8 t;
    t.lo = *reinterpret_cast<const uint4*>(lo);
    t.hi = *reinterpret_cast<const uint4*>(hi);
    return __builtin_bit_cast(v16bf, t);
}

static __device__ __forceinline__ v16bf frag_glb(const unsigned short* p) {
    const uint4* q = reinterpret_cast<const uint4*>(p);
    U32x8 t; t.lo = q[0]; t.hi = q[1];
    return __builtin_bit_cast(v16bf, t);
}

static __device__ __forceinline__ v8f vzero8() {
    v8f z;
    for (int e = 0; e < 8; ++e) z[e] = 0.0f;
    return z;
}

// ---------------------------------------------------------------------------
// K0: one-time f32 -> bf16 weight conversion into fragment-swizzled layout.
// dst linear index t = ((ct*nks + ks)*32 + lane)*16 + e
//   lane = half*16 + n ; element e -> K = ks*32 + half*16 + e ; col = ct*16 + n
// so a wave's B fragment (ct,ks) is 32 lanes x 32B, contiguous 1KB.
// ---------------------------------------------------------------------------
__global__ __launch_bounds__(256) void k0_conv_swizzle(
    const float* __restrict__ src, unsigned short* __restrict__ dst,
    int K, int Ncols)
{
    const int nks = K >> 5;
    const int total = K * Ncols;
    for (int t = blockIdx.x * 256 + threadIdx.x; t < total; t += gridDim.x * 256) {
        int e    = t & 15;
        int lane = (t >> 4) & 31;
        int ks   = (t >> 9) % nks;
        int ct   = t / (nks << 9);
        int k    = (ks << 5) + ((lane >> 4) << 4) + e;
        int col  = (ct << 4) + (lane & 15);
        dst[t] = f2bf(src[(size_t)k * Ncols + col]);
    }
}

// ---------------------------------------------------------------------------
// K1: tmid = relu(x @ W1). Block tile 128 x 512 (full width): x read ONCE.
// 512 threads = 16 waves: wave = (colhalf<<3)|rstrip -> rows rstrip*16,
// cols colhalf*256 (16 col-tiles, 16 accumulators). B direct from L2 (bf16),
// schedule pinned: 4 B fragments always in flight ahead of the WMMAs.
// ---------------------------------------------------------------------------
#define A1_STR 40   // 32 K-halfs + 8 pad (80B rows: conflict-free 16B reads)

__global__ __launch_bounds__(512) void k1_gemm_relu(
    const float* __restrict__ x, const unsigned short* __restrict__ W1b,
    float* __restrict__ tmid)
{
    __shared__ __align__(16) unsigned short Ash[128 * A1_STR]; // 10,240 B

    const int tid    = threadIdx.x;
    const int wave   = tid >> 5;
    const int lane   = tid & 31;
    const int half   = lane >> 4;
    const int l16    = lane & 15;
    const int rstrip = wave & 7;          // 8 row strips of 16
    const int ctbase = (wave >> 3) * 16;  // 2 col halves of 16 tiles
    const int row0   = blockIdx.x * 128;

    // per-lane base of this wave's B fragments: frag (ct,ks) at
    // W1b + ((ct*nks + ks)*32 + lane)*16 halfs
    const unsigned short* bbase = W1b + ((size_t)lane << 4)
                                + (((size_t)ctbase * (D_IN / 32)) << 9);

    v8f acc[16];
    for (int t = 0; t < 16; ++t) acc[t] = vzero8();

    for (int ks = 0; ks < D_IN / 32; ++ks) {
        // stage A: 128 rows x 32 cols f32 -> bf16 (1024 float4, 2 per thread)
        for (int i = 0; i < 2; ++i) {
            int q = tid + 512 * i;
            int r = q >> 3;                 // row 0..127
            int s = q & 7;                  // 4-float segment
            int grow = row0 + r;
            float4 v = make_float4(0.f, 0.f, 0.f, 0.f);
            if (grow < N_INST)
                v = *reinterpret_cast<const float4*>(
                        &x[(size_t)grow * D_IN + ks * 32 + s * 4]);
            *reinterpret_cast<uint2*>(&Ash[r * A1_STR + s * 4]) =
                make_uint2(pk2bf(v.x, v.y), pk2bf(v.z, v.w));
        }
        __syncthreads();

        const unsigned short* alo = &Ash[(rstrip * 16 + l16) * A1_STR + half * 16];
        v16bf a = frag_lds(alo, alo + 8);

        const unsigned short* bks = bbase + ((size_t)ks << 9);
        v16bf b0 = frag_glb(bks + (size_t)0 * (D_IN / 32) * 512);
        v16bf b1 = frag_glb(bks + (size_t)1 * (D_IN / 32) * 512);
        v16bf b2 = frag_glb(bks + (size_t)2 * (D_IN / 32) * 512);
        v16bf b3 = frag_glb(bks + (size_t)3 * (D_IN / 32) * 512);
#pragma unroll
        for (int t = 0; t < 16; t += 4) {
            v16bf u0 = b0, u1 = b1, u2 = b2, u3 = b3;
            if (t + 4 < 16) {
                b0 = frag_glb(bks + (size_t)(t + 4) * (D_IN / 32) * 512);
                b1 = frag_glb(bks + (size_t)(t + 5) * (D_IN / 32) * 512);
                b2 = frag_glb(bks + (size_t)(t + 6) * (D_IN / 32) * 512);
                b3 = frag_glb(bks + (size_t)(t + 7) * (D_IN / 32) * 512);
            }
            acc[t + 0] = __builtin_amdgcn_wmma_f32_16x16x32_bf16(
                false, a, false, u0, (short)0, acc[t + 0], false, false);
            acc[t + 1] = __builtin_amdgcn_wmma_f32_16x16x32_bf16(
                false, a, false, u1, (short)0, acc[t + 1], false, false);
            acc[t + 2] = __builtin_amdgcn_wmma_f32_16x16x32_bf16(
                false, a, false, u2, (short)0, acc[t + 2], false, false);
            acc[t + 3] = __builtin_amdgcn_wmma_f32_16x16x32_bf16(
                false, a, false, u3, (short)0, acc[t + 3], false, false);
        }
        // pin the schedule for this region: A frag, 4 B frags in flight,
        // then 1 WMMA per 1 prefetched fragment, drain at the end.
        SGB(SGB_DSREAD, 2, 0);       // A fragment (2x ds_load_b128)
        SGB(SGB_VMREAD, 8, 0);       // B frags 0..3 (8x global_load_b128)
#pragma unroll
        for (int i = 0; i < 12; ++i) {
            SGB(SGB_WMMA, 1, 0);     // compute tile i
            SGB(SGB_VMREAD, 2, 0);   // prefetch tile i+4
        }
        SGB(SGB_WMMA, 4, 0);         // drain last 4 tiles
        __syncthreads();
    }

    const int rbase = row0 + rstrip * 16 + (half ? 8 : 0);
    for (int t = 0; t < 16; ++t) {
        int col = (ctbase + t) * 16 + l16;
        for (int r = 0; r < 8; ++r) {
            int row = rbase + r;
            if (row < N_INST) {
                float v = acc[t][r];
                tmid[(size_t)row * DD + col] = v > 0.0f ? v : 0.0f;
            }
        }
    }
}

// ---------------------------------------------------------------------------
// K2: per 16 rows: scores = tanh(T @ V) @ w_att ; c0/c1 = T @ Wc (f32)
// A staged once in LDS; B fragments direct from swizzled bf16 V in L2.
// Distance-1 software pipeline across k-steps, pinned with sched groups.
// ---------------------------------------------------------------------------
#define A2_STR 520  // 512 + 8 pad halfs

__global__ __launch_bounds__(256) void k2_scores(
    const float* __restrict__ tmid, const unsigned short* __restrict__ Vb,
    const float* __restrict__ w_att, const float* __restrict__ Wc,
    float* __restrict__ scores, float* __restrict__ c0, float* __restrict__ c1)
{
    __shared__ __align__(16) unsigned short Ash[16 * A2_STR];   // 16,640 B
    __shared__ float red0[256];
    __shared__ float red1[256];
    __shared__ float score_sh[16];

    const int tid  = threadIdx.x;
    const int wave = tid >> 5;
    const int lane = tid & 31;
    const int half = lane >> 4;
    const int l16  = lane & 15;
    const int r0   = blockIdx.x * 16;   // 6250 blocks exactly

    // stage A (f32 -> bf16) + f32 partial c0/c1
    {
        int row = tid >> 4;            // 0..15
        int seg = tid & 15;            // cols seg*32..+31
        int grow = r0 + row;
        float p0 = 0.f, p1 = 0.f;
        for (int i = 0; i < 8; ++i) {
            int k = seg * 32 + i * 4;
            float4 v = *reinterpret_cast<const float4*>(&tmid[(size_t)grow * DD + k]);
            *reinterpret_cast<uint2*>(&Ash[row * A2_STR + k]) =
                make_uint2(pk2bf(v.x, v.y), pk2bf(v.z, v.w));
            p0 += v.x * Wc[(k + 0) * 2] + v.y * Wc[(k + 1) * 2]
                + v.z * Wc[(k + 2) * 2] + v.w * Wc[(k + 3) * 2];
            p1 += v.x * Wc[(k + 0) * 2 + 1] + v.y * Wc[(k + 1) * 2 + 1]
                + v.z * Wc[(k + 2) * 2 + 1] + v.w * Wc[(k + 3) * 2 + 1];
        }
        red0[tid] = p0; red1[tid] = p1;
        if (tid < 16) score_sh[tid] = 0.0f;
        __syncthreads();
        if (seg == 0) {
            float s0 = 0.f, s1 = 0.f;
            for (int i = 0; i < 16; ++i) { s0 += red0[row * 16 + i]; s1 += red1[row * 16 + i]; }
            c0[grow] = s0; c1[grow] = s1;
        }
    }

    v8f acc[4];
    for (int t = 0; t < 4; ++t) acc[t] = vzero8();

    // per-lane base for this wave's 4 col-tiles; frag (ct,ks) at
    // Vb + ((ct*nks + ks)*32 + lane)*16 halfs, nks = 16
    const unsigned short* bbase = Vb + ((size_t)lane << 4)
                                + (((size_t)(wave * 4) * (DD / 32)) << 9);
    const unsigned short* a0p = &Ash[l16 * A2_STR + half * 16];

    // Ash is read-only from here; no further barriers needed in the k-loop.
    v16bf a_cur = frag_lds(a0p, a0p + 8);
    v16bf b0 = frag_glb(bbase + (size_t)0 * (DD / 32) * 512);
    v16bf b1 = frag_glb(bbase + (size_t)1 * (DD / 32) * 512);
    v16bf b2 = frag_glb(bbase + (size_t)2 * (DD / 32) * 512);
    v16bf b3 = frag_glb(bbase + (size_t)3 * (DD / 32) * 512);
#pragma unroll
    for (int ks = 0; ks < DD / 32; ++ks) {
        v16bf au = a_cur, u0 = b0, u1 = b1, u2 = b2, u3 = b3;
        if (ks + 1 < DD / 32) {
            const unsigned short* an = a0p + (ks + 1) * 32;
            a_cur = frag_lds(an, an + 8);
            const unsigned short* bn = bbase + ((size_t)(ks + 1) << 9);
            b0 = frag_glb(bn + (size_t)0 * (DD / 32) * 512);
            b1 = frag_glb(bn + (size_t)1 * (DD / 32) * 512);
            b2 = frag_glb(bn + (size_t)2 * (DD / 32) * 512);
            b3 = frag_glb(bn + (size_t)3 * (DD / 32) * 512);
        }
        acc[0] = __builtin_amdgcn_wmma_f32_16x16x32_bf16(
            false, au, false, u0, (short)0, acc[0], false, false);
        acc[1] = __builtin_amdgcn_wmma_f32_16x16x32_bf16(
            false, au, false, u1, (short)0, acc[1], false, false);
        acc[2] = __builtin_amdgcn_wmma_f32_16x16x32_bf16(
            false, au, false, u2, (short)0, acc[2], false, false);
        acc[3] = __builtin_amdgcn_wmma_f32_16x16x32_bf16(
            false, au, false, u3, (short)0, acc[3], false, false);
        // per-iteration pin: issue next iteration's loads before this
        // iteration's WMMAs (distance-1 pipeline)
        SGB(SGB_DSREAD, 2, 0);
        SGB(SGB_VMREAD, 8, 0);
        SGB(SGB_WMMA, 4, 0);
    }

    // epilogue: tanh, dot with w_att, reduce over columns
    float ps[8];
    for (int r = 0; r < 8; ++r) ps[r] = 0.0f;
    for (int t = 0; t < 4; ++t) {
        int col = (wave * 4 + t) * 16 + l16;
        float wa = w_att[col];
        for (int r = 0; r < 8; ++r) ps[r] += tanhf(acc[t][r]) * wa;
    }
    for (int r = 0; r < 8; ++r) {
        float v = ps[r];
        v += __shfl_xor(v, 1, 32);
        v += __shfl_xor(v, 2, 32);
        v += __shfl_xor(v, 4, 32);
        v += __shfl_xor(v, 8, 32);
        if (l16 == 0) atomicAdd(&score_sh[(half ? 8 : 0) + r], v);
    }
    __syncthreads();
    if (tid < 16) scores[r0 + tid] = score_sh[tid];
}

// ---------------------------------------------------------------------------
// K3: global softmax stats (two-level max, then two-level sum-exp)
// ---------------------------------------------------------------------------
__global__ __launch_bounds__(256) void k3a_max(const float* __restrict__ s,
                                               float* __restrict__ pm)
{
    __shared__ float red[256];
    int tid = threadIdx.x;
    float m = -__builtin_inff();
    for (int i = blockIdx.x * 256 + tid; i < N_INST; i += 256 * 256)
        m = fmaxf(m, s[i]);
    red[tid] = m; __syncthreads();
    for (int w = 128; w > 0; w >>= 1) {
        if (tid < w) red[tid] = fmaxf(red[tid], red[tid + w]);
        __syncthreads();
    }
    if (tid == 0) pm[blockIdx.x] = red[0];
}

__global__ __launch_bounds__(256) void k3b_max(const float* __restrict__ pm,
                                               float* __restrict__ stats)
{
    __shared__ float red[256];
    int tid = threadIdx.x;
    red[tid] = pm[tid]; __syncthreads();
    for (int w = 128; w > 0; w >>= 1) {
        if (tid < w) red[tid] = fmaxf(red[tid], red[tid + w]);
        __syncthreads();
    }
    if (tid == 0) stats[0] = red[0];
}

__global__ __launch_bounds__(256) void k3c_sum(const float* __restrict__ s,
                                               const float* __restrict__ stats,
                                               float* __restrict__ psum)
{
    __shared__ float red[256];
    int tid = threadIdx.x;
    float m = stats[0];
    float acc = 0.0f;
    for (int i = blockIdx.x * 256 + tid; i < N_INST; i += 256 * 256)
        acc += expf(s[i] - m);
    red[tid] = acc; __syncthreads();
    for (int w = 128; w > 0; w >>= 1) {
        if (tid < w) red[tid] += red[tid + w];
        __syncthreads();
    }
    if (tid == 0) psum[blockIdx.x] = red[0];
}

__global__ __launch_bounds__(256) void k3d_sum(const float* __restrict__ psum,
                                               float* __restrict__ stats)
{
    __shared__ float red[256];
    int tid = threadIdx.x;
    red[tid] = psum[tid]; __syncthreads();
    for (int w = 128; w > 0; w >>= 1) {
        if (tid < w) red[tid] += red[tid + w];
        __syncthreads();
    }
    if (tid == 0) stats[1] = red[0];
}

// ---------------------------------------------------------------------------
// K4: one block per group: re-softmax of AA, preds, argmax/argmin, pseudo copy
// ---------------------------------------------------------------------------
__global__ __launch_bounds__(256) void k4_groups(
    const float* __restrict__ tmid, const int* __restrict__ idx,
    const float* __restrict__ scores, const float* __restrict__ c0,
    const float* __restrict__ c1, const float* __restrict__ bc,
    const float* __restrict__ stats, float* __restrict__ out)
{
    __shared__ float rf[256];
    __shared__ float r_a0[256], r_a1[256], r_vx[256], r_vn[256];
    __shared__ int   r_ix[256], r_in[256];
    __shared__ int   sel[2];

    const int g   = blockIdx.x;
    const int tid = threadIdx.x;
    const int* gidx = &idx[g * GS];
    const float m = stats[0];
    const float invZ = 1.0f / stats[1];

    // pass 1: max of AA over group
    float lm = -__builtin_inff();
    for (int j = tid; j < GS; j += 256) {
        float aa = expf(scores[gidx[j]] - m) * invZ;
        lm = fmaxf(lm, aa);
    }
    rf[tid] = lm; __syncthreads();
    for (int w = 128; w > 0; w >>= 1) {
        if (tid < w) rf[tid] = fmaxf(rf[tid], rf[tid + w]);
        __syncthreads();
    }
    const float gmax = rf[0];
    __syncthreads();

    // pass 1b: sum exp(AA - gmax)
    float lsum = 0.0f;
    for (int j = tid; j < GS; j += 256) {
        float aa = expf(scores[gidx[j]] - m) * invZ;
        lsum += expf(aa - gmax);
    }
    rf[tid] = lsum; __syncthreads();
    for (int w = 128; w > 0; w >>= 1) {
        if (tid < w) rf[tid] += rf[tid + w];
        __syncthreads();
    }
    const float invgz = 1.0f / rf[0];
    __syncthreads();

    // pass 2: preds + first-occurrence argmax/argmin of tAA*(c1-c0)
    float a0 = 0.f, a1 = 0.f;
    float bmax = -__builtin_inff(); int bimax = 0x7FFFFFFF;
    float bmin =  __builtin_inff(); int bimin = 0x7FFFFFFF;
    for (int j = tid; j < GS; j += 256) {
        int gi = gidx[j];
        float aa = expf(scores[gi] - m) * invZ;
        float ta = expf(aa - gmax) * invgz;
        float v0 = c0[gi], v1 = c1[gi];
        a0 += ta * v0;
        a1 += ta * v1;
        float d = ta * (v1 - v0);
        if (d > bmax) { bmax = d; bimax = j; }
        if (d < bmin) { bmin = d; bimin = j; }
    }
    r_a0[tid] = a0; r_a1[tid] = a1;
    r_vx[tid] = bmax; r_ix[tid] = bimax;
    r_vn[tid] = bmin; r_in[tid] = bimin;
    __syncthreads();

    if (tid == 0) {
        float s0 = 0.f, s1 = 0.f;
        float vx = -__builtin_inff(); int ix = 0x7FFFFFFF;
        float vn =  __builtin_inff(); int in_ = 0x7FFFFFFF;
        for (int t = 0; t < 256; ++t) {
            s0 += r_a0[t]; s1 += r_a1[t];
            if (r_vx[t] > vx || (r_vx[t] == vx && r_ix[t] < ix)) { vx = r_vx[t]; ix = r_ix[t]; }
            if (r_vn[t] < vn || (r_vn[t] == vn && r_in[t] < in_)) { vn = r_vn[t]; in_ = r_in[t]; }
        }
        out[g * NC + 0] = s0 + bc[0];
        out[g * NC + 1] = s1 + bc[1];
        sel[0] = ix; sel[1] = in_;
    }
    __syncthreads();

    // faithful quirk: local argmax/argmin indices used as GLOBAL tmid rows
    const int imax = sel[0], imin = sel[1];
    float* pf = &out[NG * NC + g * 2 * DD];
    for (int c = tid; c < DD; c += 256) {
        pf[c]      = tmid[(size_t)imax * DD + c];
        pf[DD + c] = tmid[(size_t)imin * DD + c];
    }
}

// ---------------------------------------------------------------------------
extern "C" void kernel_launch(void* const* d_in, const int* in_sizes, int n_in,
                              void* d_out, int out_size, void* d_ws, size_t ws_size,
                              hipStream_t stream) {
    const float* x     = (const float*)d_in[0];
    const int*   idx   = (const int*)  d_in[1];
    const float* W1    = (const float*)d_in[2];
    const float* Vm    = (const float*)d_in[3];
    const float* w_att = (const float*)d_in[4];
    const float* Wc    = (const float*)d_in[5];
    const float* bc    = (const float*)d_in[6];
    float* out = (float*)d_out;

    float* ws     = (float*)d_ws;
    float* tmid   = ws;                                  // N*512 f32
    float* scores = tmid + (size_t)N_INST * DD;          // N
    float* c0     = scores + N_INST;                     // N
    float* c1     = c0 + N_INST;                         // N
    float* pm     = c1 + N_INST;                         // 256
    float* psum   = pm + 256;                            // 256
    float* stats  = psum + 256;                          // 32 (padded, keeps 16B alignment)
    unsigned short* W1b = (unsigned short*)(stats + 32); // 1024*512 bf16 (swizzled)
    unsigned short* Vb  = W1b + (size_t)D_IN * DD;       // 512*512 bf16 (swizzled)

    k0_conv_swizzle<<<dim3(512), dim3(256), 0, stream>>>(W1, W1b, D_IN, DD);
    k0_conv_swizzle<<<dim3(256), dim3(256), 0, stream>>>(Vm, Vb, DD, DD);
    k1_gemm_relu<<<dim3((N_INST + 127) / 128), dim3(512), 0, stream>>>(x, W1b, tmid);
    k2_scores<<<dim3(N_INST / 16), dim3(256), 0, stream>>>(tmid, Vb, w_att, Wc, scores, c0, c1);
    k3a_max<<<dim3(256), dim3(256), 0, stream>>>(scores, pm);
    k3b_max<<<dim3(1), dim3(256), 0, stream>>>(pm, stats);
    k3c_sum<<<dim3(256), dim3(256), 0, stream>>>(scores, stats, psum);
    k3d_sum<<<dim3(1), dim3(256), 0, stream>>>(psum, stats);
    k4_groups<<<dim3(NG), dim3(256), 0, stream>>>(tmid, idx, scores, c0, c1, bc, stats, out);
}
